// RGCNLayer_11046655885897
// MI455X (gfx1250) — matrix operand order repaired and checked by reading, not compiled, not run on hardware
//
#include <hip/hip_runtime.h>

typedef __attribute__((ext_vector_type(16))) __bf16 v16bf;
typedef __attribute__((ext_vector_type(8)))  __bf16 v8bf;
typedef __attribute__((ext_vector_type(8)))  float  v8f;
typedef __attribute__((ext_vector_type(4)))  float  v4f;

#define NN 4096   // nodes
#define FF 128    // in/out features

// ---------------------------------------------------------------------------
// Kernel 1: per-node transform (memory-bound batched matvecs)
//   support[k] = sum_j input[n,j]   * weights[n,j,k]
//   t[j]       = sum_k rel[n,j,k]   * support[k]
// All global streaming uses b128 loads (512 B per wave per instruction) to
// keep enough bytes in flight for 23.3 TB/s.
// Output written TRANSPOSED in bf16: tmpT[f][n] so the GEMM's B fragments are
// per-lane K-contiguous (global_load_b128 from L2).
// ---------------------------------------------------------------------------
__global__ __launch_bounds__(128) void rgcn_node_transform(
    const float* __restrict__ input,    // [NN, FF]
    const float* __restrict__ rel,      // [NN, FF, FF]
    const float* __restrict__ weights,  // [NN, FF, FF]
    __bf16* __restrict__ tmpT)          // [FF, NN]
{
    __shared__ __align__(16) float in_s[FF];
    __shared__ v4f              part[4][32];   // partial support [row-group][col-chunk]
    __shared__ __align__(16) float sup_s[FF];

    const int n = blockIdx.x;
    const int t = threadIdx.x;          // 0..127

    in_s[t] = input[(size_t)n * FF + t];
    __syncthreads();

    // ---- support matvec: thread t = (g,c) owns columns 4c..4c+3, rows j=g,g+4,...
    // loads are b128, contiguous 512 B across the 32 c-threads of a row
    const int c = t & 31;               // float4 column chunk
    const int g = t >> 5;               // j row-group
    const v4f* W4 = (const v4f*)(weights + (size_t)n * FF * FF);
    v4f acc = {0.f, 0.f, 0.f, 0.f};
#pragma unroll 8
    for (int j = g; j < FF; j += 4) {
        const float x = in_s[j];
        const v4f   w = W4[j * 32 + c];
        acc.x = fmaf(x, w.x, acc.x);
        acc.y = fmaf(x, w.y, acc.y);
        acc.z = fmaf(x, w.z, acc.z);
        acc.w = fmaf(x, w.w, acc.w);
    }
    part[g][c] = acc;
    __syncthreads();

    // ---- 4-way reduction across row-groups: thread t owns support[t]
    {
        const float* pf = (const float*)part;          // viewed as [4][128]
        sup_s[t] = (pf[t] + pf[FF + t]) + (pf[2 * FF + t] + pf[3 * FF + t]);
    }
    __syncthreads();

    // ---- relational combine: thread t owns output row t; b128 streaming of rel
    const v4f* R = (const v4f*)(rel + (size_t)n * FF * FF + (size_t)t * FF);
    const v4f* S = (const v4f*)sup_s;
    float acc2 = 0.f;
#pragma unroll 8
    for (int q = 0; q < FF / 4; ++q) {
        const v4f r = R[q];
        const v4f s = S[q];
        acc2 = fmaf(r.x, s.x, acc2);
        acc2 = fmaf(r.y, s.y, acc2);
        acc2 = fmaf(r.z, s.z, acc2);
        acc2 = fmaf(r.w, s.w, acc2);
    }

    tmpT[(size_t)t * NN + n] = (__bf16)acc2;
}

// ---------------------------------------------------------------------------
// Kernel 2: out = adjacency @ tmp + bias  via V_WMMA_F32_16X16X32_BF16
//   grid  = 256 blocks (16-row M slabs)
//   block = 256 threads = 8 waves; wave w computes N-tile w (cols 16w..16w+15)
//   A slab (16 x 128 per K-chunk) staged f32->bf16 in LDS, shared by all waves
//   B read from L2-resident bf16 tmpT (transposed), two b128 loads per WMMA
// ---------------------------------------------------------------------------
__global__ __launch_bounds__(256) void rgcn_aggregate_wmma(
    const float*  __restrict__ adj,   // [NN, NN] f32
    const __bf16* __restrict__ tmpT,  // [FF, NN] bf16
    const float*  __restrict__ bias,  // [FF]
    float*        __restrict__ out)   // [NN, FF]
{
    // row pad 136 bf16 (272 B): fragment ds reads hit distinct banks (4m mod 64)
    __shared__ __bf16 As[16][136];

    const int row_base = blockIdx.x * 16;
    const int tid  = threadIdx.x;
    const int wave = tid >> 5;
    const int lane = tid & 31;
    const int h    = lane >> 4;        // half-wave select
    const int mn   = lane & 15;        // A row (m) / B,C col (n)
    const int col_base = wave * 16;

    v8f c = {};                        // f32 accumulators (16x16 C tile)

    // A staging map: 256 threads cover 16 rows x 128 K, 8 floats each
    const int srow = tid >> 4;         // 0..15
    const int scol = (tid & 15) * 8;   // 0,8,...,120

    const __bf16* bcol = tmpT + (size_t)(col_base + mn) * NN;

    for (int k0 = 0; k0 < NN; k0 += 128) {
        // ---- stage + convert A tile (16 x 128 f32 -> bf16 LDS) ----
        const float* ap = adj + (size_t)(row_base + srow) * NN + k0 + scol;
        v4f a0 = *(const v4f*)(ap);
        v4f a1 = *(const v4f*)(ap + 4);
        v8bf pk;
        pk[0] = (__bf16)a0.x; pk[1] = (__bf16)a0.y;
        pk[2] = (__bf16)a0.z; pk[3] = (__bf16)a0.w;
        pk[4] = (__bf16)a1.x; pk[5] = (__bf16)a1.y;
        pk[6] = (__bf16)a1.z; pk[7] = (__bf16)a1.w;
        *(v8bf*)&As[srow][scol] = pk;
        __syncthreads();

        // ---- 4 WMMAs per chunk (K = 32 each) ----
#pragma unroll
        for (int kk = 0; kk < 128; kk += 32) {
            union { v16bf v; v8bf p[2]; } ua, ub;
            // A 16x32 bf16 fragment: lane holds row mn,
            //   K in [kk+16h .. kk+16h+15]
            ua.p[0] = *(const v8bf*)&As[mn][kk + 16 * h];
            ua.p[1] = *(const v8bf*)&As[mn][kk + 16 * h + 8];
            // B 32x16 bf16 fragment: lane holds col mn, K in [kk+16h .. +15]
            const __bf16* bp = bcol + k0 + kk + 16 * h;
            ub.p[0] = *(const v8bf*)(bp);
            ub.p[1] = *(const v8bf*)(bp + 8);

            c = __builtin_amdgcn_wmma_f32_16x16x32_bf16(
                    /*neg_a=*/false, ua.v, /*neg_b=*/false, ub.v,
                    /*c_mod=*/(short)0, c, /*reuse_a=*/false, /*reuse_b=*/false);
        }
        __syncthreads();
    }

    // ---- epilogue: bias + store (C/D layout: VGPR v -> m = v + 8h) ----
    const int col  = col_base + mn;
    const float bb = bias[col];
    const int row0 = row_base + h * 8;
#pragma unroll
    for (int v = 0; v < 8; ++v)
        out[(size_t)(row0 + v) * FF + col] = c[v] + bb;
}

// ---------------------------------------------------------------------------
extern "C" void kernel_launch(void* const* d_in, const int* in_sizes, int n_in,
                              void* d_out, int out_size, void* d_ws, size_t ws_size,
                              hipStream_t stream) {
    const float* input = (const float*)d_in[0];   // [4096,128]
    const float* adj   = (const float*)d_in[1];   // [4096,4096]
    const float* rel   = (const float*)d_in[2];   // [4096,128,128]
    const float* wts   = (const float*)d_in[3];   // [4096,128,128]
    const float* bias  = (const float*)d_in[4];   // [128]
    float* out = (float*)d_out;                   // [4096,128]

    __bf16* tmpT = (__bf16*)d_ws;                 // [128,4096] bf16 = 1 MB

    rgcn_node_transform<<<NN, 128, 0, stream>>>(input, rel, wts, tmpT);
    rgcn_aggregate_wmma<<<NN / 16, 256, 0, stream>>>(adj, tmpT, bias, out);
}
